// HyperGCNConv_81020263071821
// MI455X (gfx1250) — matrix operand order repaired and checked by reading, not compiled, not run on hardware
//
#include <hip/hip_runtime.h>
#include <hip/hip_bf16.h>

typedef __attribute__((ext_vector_type(16))) __bf16 v16bf;
typedef __attribute__((ext_vector_type(8)))  float  v8f;

#define DIN  128
#define DOUT 128
#define KHE  8                 // hyperedge size
#define MED_W (1.0f / 13.0f)   // 1/(2K-3)

// ---------------------------------------------------------------- utilities
__device__ __forceinline__ float wave_reduce_add(float v) {
#pragma unroll
    for (int off = 16; off > 0; off >>= 1)
        v += __shfl_xor(v, off, 32);
    return v;
}

// ---------------------------------------------------------------- k0: deg = 1
__global__ void __launch_bounds__(256)
k0_init_deg(float* __restrict__ deg, int n) {
    int i = blockIdx.x * 256 + threadIdx.x;
    if (i < n) deg[i] = 1.0f;
}

// ---------------------------------------------------------------- k1: Y = X@W + b  (bf16 WMMA, f32 accum)
// Persistent blocks, grid-stride over 16-row M tiles. 8 waves/block, wave w
// owns cols [16w,16w+16). W staged once into LDS in WMMA B-fragment order
// (each lane's fragment = 32 contiguous bytes), hoisted into registers for
// the whole tile loop. Full tiles store through one base pointer with
// immediate offsets (no per-row bound checks in the hot path).
__global__ void __launch_bounds__(256)
k1_gemm_wmma(const float* __restrict__ X, const float* __restrict__ W,
             const float* __restrict__ bias, float* __restrict__ Y,
             int N, int ntiles) {
    // WB[kgroup][col][half*16 + i] : kgroup = k>>5, half = (k>>3)&1, i = ((k>>4)&1)*8 + (k&7)
    __shared__ __bf16 WB[4 * 128 * 32];        // 32 KB

    for (int idx = threadIdx.x; idx < DIN * DOUT; idx += 256) {
        const int k = idx >> 7;                // K index (row of W)
        const int c = idx & 127;               // N index (col of W)
        WB[((k >> 5) * 128 + c) * 32 + ((k >> 3) & 1) * 16 + ((k >> 4) & 1) * 8 + (k & 7)] =
            (__bf16)W[idx];
    }
    __syncthreads();

    const int lane = threadIdx.x & 31;
    const int wave = threadIdx.x >> 5;
    const int half = lane >> 4;        // 0: lanes 0-15, 1: lanes 16-31
    const int l16  = lane & 15;
    const int col  = wave * 16 + l16;
    const int kb   = half * 8;         // A-fragment K base for this lane half

    // B fragments: loaded once, kept in registers (4 x 8 VGPRs)
    v16bf bfrag[4];
#pragma unroll
    for (int kg = 0; kg < 4; ++kg)
        bfrag[kg] = *(const v16bf*)(WB + (kg * 128 + col) * 32 + half * 16);

    const float bv = bias[col];

    for (int tile = blockIdx.x; tile < ntiles; tile += gridDim.x) {
        const int m0 = tile * 16;
        int row = m0 + l16;
        if (row >= N) row = N - 1;     // pad rows duplicate last row; stores guarded
        const float* xr = X + (size_t)row * DIN + kb;

        v8f acc = {};
#pragma unroll
        for (int kg = 0; kg < 4; ++kg) {
            // A fragment: half 0 -> K = 32kg+{0..7,16..23}; half 1 -> +8
            const float4* xp = (const float4*)(xr + kg * 32);
            float4 x0 = xp[0], x1 = xp[1];     // K = kb .. kb+7
            float4 x2 = xp[4], x3 = xp[5];     // K = kb+16 .. kb+23
            float ax[16] = { x0.x, x0.y, x0.z, x0.w, x1.x, x1.y, x1.z, x1.w,
                             x2.x, x2.y, x2.z, x2.w, x3.x, x3.y, x3.z, x3.w };
            v16bf a;
#pragma unroll
            for (int i = 0; i < 16; ++i) a[i] = (__bf16)ax[i];
            acc = __builtin_amdgcn_wmma_f32_16x16x32_bf16(
                      false, a, false, bfrag[kg], (short)0, acc, false, false);
        }

        // D: VGPR j -> M = m0 + 8*half + j, N = col
        float* yo = Y + (size_t)(m0 + half * 8) * DOUT + col;
        if (m0 + 16 <= N) {            // full tile: base + immediate offsets
#pragma unroll
            for (int j = 0; j < 8; ++j)
                yo[(size_t)j * DOUT] = acc[j] + bv;
        } else {                        // ragged last tile
#pragma unroll
            for (int j = 0; j < 8; ++j)
                if (m0 + half * 8 + j < N) yo[(size_t)j * DOUT] = acc[j] + bv;
        }

        // prefetch next tile's A rows (global_prefetch_b8)
        const int nt = tile + gridDim.x;
        if (nt < ntiles) {
            int nr = nt * 16 + l16;
            if (nr >= N) nr = N - 1;
            __builtin_prefetch(X + (size_t)nr * DIN + kb, 0, 1);
        }
    }
}

// ---------------------------------------------------------------- k2: argmax-distance pair + degree atomics
// one wave32 per hyperedge; lane owns dims [4*lane, 4*lane+4)
__global__ void __launch_bounds__(256)
k2_edge_deg(const float* __restrict__ Y, const int* __restrict__ vertex,
            float* __restrict__ deg, int* __restrict__ uvbuf, int M) {
    const int lane = threadIdx.x & 31;
    const int m    = blockIdx.x * 8 + (threadIdx.x >> 5);
    if (m >= M) return;

    int vload = 0;
    if (lane < KHE) vload = vertex[(size_t)m * KHE + lane];
    int vk[KHE];
#pragma unroll
    for (int k = 0; k < KHE; ++k) vk[k] = __shfl(vload, k, 32);

    float f[KHE][4];
#pragma unroll
    for (int r = 0; r < KHE; ++r) {
        const float4 t = *(const float4*)(Y + (size_t)vk[r] * DOUT + lane * 4);
        f[r][0] = t.x; f[r][1] = t.y; f[r][2] = t.z; f[r][3] = t.w;
    }

    float sq[KHE];
#pragma unroll
    for (int r = 0; r < KHE; ++r) {
        float p = f[r][0]*f[r][0] + f[r][1]*f[r][1] + f[r][2]*f[r][2] + f[r][3]*f[r][3];
        sq[r] = wave_reduce_add(p);
    }

    // argmax over d2 (row-major, first occurrence). Symmetric, zero diag:
    // enumerate r<s pairs in (r,s) order with strict '>' => matches jnp.argmax.
    float best = 0.0f; int bi = 0;
#pragma unroll
    for (int r = 0; r < KHE; ++r) {
#pragma unroll
        for (int s = r + 1; s < KHE; ++s) {
            float p = f[r][0]*f[s][0] + f[r][1]*f[s][1] + f[r][2]*f[s][2] + f[r][3]*f[s][3];
            float g = wave_reduce_add(p);
            float d = sq[r] + sq[s] - 2.0f * g;
            if (d > best) { best = d; bi = r * KHE + s; }
        }
    }
    const int ui = bi >> 3, vi = bi & 7;
    const int cnt = KHE - ((ui == vi) ? 1 : 2);   // mediator count

    // degree: pos==ui/vi get w*(1+cnt) each; mediators get 2w
    if (lane < KHE) {
        const int k = lane;
        float fac = 0.0f;
        if (k == ui) fac += 1.0f + (float)cnt;
        if (k == vi) fac += 1.0f + (float)cnt;
        if (k != ui && k != vi) fac += 2.0f;
        atomicAdd(&deg[vk[k]], MED_W * fac);
    }
    if (lane == 0) uvbuf[m] = ui | (vi << 4);
}

// ---------------------------------------------------------------- k3a: dinv = rsqrt(deg)
__global__ void __launch_bounds__(256)
k3a_rsqrt(const float* __restrict__ deg, float* __restrict__ dinv, int n) {
    int i = blockIdx.x * 256 + threadIdx.x;
    if (i < n) dinv[i] = rsqrtf(deg[i]);
}

// ---------------------------------------------------------------- k3b: out = Y * dinv (self-loop term Xs)
__global__ void __launch_bounds__(256)
k3b_selfloop(const float* __restrict__ Y, const float* __restrict__ dinv,
             float* __restrict__ out, int nvec) {     // nvec = N*32 float4s
    int i = blockIdx.x * 256 + threadIdx.x;
    if (i >= nvec) return;
    const float d = dinv[i >> 5];
    float4 y = ((const float4*)Y)[i];
    ((float4*)out)[i] = make_float4(y.x * d, y.y * d, y.z * d, y.w * d);
}

// ---------------------------------------------------------------- k4: fused symmetric scatter (atomics)
__global__ void __launch_bounds__(256)
k4_scatter(const float* __restrict__ Y, const int* __restrict__ vertex,
           const float* __restrict__ dinv, const int* __restrict__ uvbuf,
           float* __restrict__ out, int M) {
    const int lane = threadIdx.x & 31;
    const int m    = blockIdx.x * 8 + (threadIdx.x >> 5);
    if (m >= M) return;

    const int uv = uvbuf[m];
    const int ui = uv & 15, vi = (uv >> 4) & 15;

    int vload = 0;
    if (lane < KHE) vload = vertex[(size_t)m * KHE + lane];
    int vk[KHE];
#pragma unroll
    for (int k = 0; k < KHE; ++k) vk[k] = __shfl(vload, k, 32);

    float xs[KHE][4];
#pragma unroll
    for (int r = 0; r < KHE; ++r) {
        const float dr = dinv[vk[r]];
        const float4 t = *(const float4*)(Y + (size_t)vk[r] * DOUT + lane * 4);
        xs[r][0] = t.x * dr; xs[r][1] = t.y * dr;
        xs[r][2] = t.z * dr; xs[r][3] = t.w * dr;
    }

    // Xs_u, Xs_v, mediator sum (static indexing only — predicated accumulation)
    float xu[4] = {0,0,0,0}, xv[4] = {0,0,0,0}, S[4] = {0,0,0,0};
#pragma unroll
    for (int k = 0; k < KHE; ++k) {
        const bool isu = (k == ui), isv = (k == vi);
#pragma unroll
        for (int j = 0; j < 4; ++j) {
            xu[j] += isu ? xs[k][j] : 0.0f;
            xv[j] += isv ? xs[k][j] : 0.0f;
            S[j]  += (!isu && !isv) ? xs[k][j] : 0.0f;
        }
    }

#pragma unroll
    for (int k = 0; k < KHE; ++k) {
        const bool isu = (k == ui), isv = (k == vi);
        float* dst = out + (size_t)vk[k] * DOUT + lane * 4;
#pragma unroll
        for (int j = 0; j < 4; ++j) {
            float a = 0.0f;
            if (isu) a += xv[j] + S[j];
            if (isv) a += xu[j] + S[j];
            if (!isu && !isv) a += xu[j] + xv[j];
            atomicAdd(dst + j, MED_W * a);
        }
    }
}

// ---------------------------------------------------------------- k5: out = relu(out * dinv)
__global__ void __launch_bounds__(256)
k5_finalize(float* __restrict__ out, const float* __restrict__ dinv, int nvec) {
    int i = blockIdx.x * 256 + threadIdx.x;
    if (i >= nvec) return;
    const float d = dinv[i >> 5];
    float4 v = ((float4*)out)[i];
    v.x = fmaxf(v.x * d, 0.0f);
    v.y = fmaxf(v.y * d, 0.0f);
    v.z = fmaxf(v.z * d, 0.0f);
    v.w = fmaxf(v.w * d, 0.0f);
    ((float4*)out)[i] = v;
}

// ---------------------------------------------------------------- launch
extern "C" void kernel_launch(void* const* d_in, const int* in_sizes, int n_in,
                              void* d_out, int out_size, void* d_ws, size_t ws_size,
                              hipStream_t stream) {
    const float* X      = (const float*)d_in[0];
    const int*   vertex = (const int*)  d_in[1];
    /* d_in[2] = edges (repeat(arange(M),K)) — implied by layout, unused */
    const float* W      = (const float*)d_in[3];
    const float* bias   = (const float*)d_in[4];
    float*       out    = (float*)d_out;

    const int N = in_sizes[0] / DIN;     // 100000
    const int M = in_sizes[1] / KHE;     // 50000

    char*  ws   = (char*)d_ws;
    float* Y    = (float*)(ws);
    float* deg  = (float*)(ws + (size_t)N * DOUT * 4);
    float* dinv = (float*)(ws + (size_t)N * DOUT * 4 + (size_t)N * 4);
    int*   uv   = (int*)  (ws + (size_t)N * DOUT * 4 + (size_t)N * 8);

    const int nvec   = N * (DOUT / 4);   // float4 count
    const int ntiles = (N + 15) / 16;
    const int gblk   = (ntiles < 1280) ? ntiles : 1280;   // persistent blocks

    k0_init_deg <<<(N + 255) / 256, 256, 0, stream>>>(deg, N);
    k1_gemm_wmma<<<gblk,            256, 0, stream>>>(X, W, bias, Y, N, ntiles);
    k2_edge_deg <<<(M + 7) / 8,     256, 0, stream>>>(Y, vertex, deg, uv, M);
    k3a_rsqrt   <<<(N + 255) / 256, 256, 0, stream>>>(deg, dinv, N);
    k3b_selfloop<<<(nvec + 255)/256,256, 0, stream>>>(Y, dinv, out, nvec);
    k4_scatter  <<<(M + 7) / 8,     256, 0, stream>>>(Y, vertex, dinv, uv, out, M);
    k5_finalize <<<(nvec + 255)/256,256, 0, stream>>>(out, dinv, nvec);
}
